// HeteroGNN_89996744720459
// MI455X (gfx1250) — compile-verified, matrix operand order checked by reading, not run on hardware
//
#include <hip/hip_runtime.h>
#include <hip/hip_bf16.h>
#include <math.h>

#define N_NODES   50000
#define M_PAD     50048            // ceil(N/128)*128 — padded GEMM row count
#define D_IN      128
#define HID_C     128
#define OUT_C     64
#define NEDGE     800000
#define NEG_SLOPE 0.2f

typedef __attribute__((ext_vector_type(16))) __bf16 bf16x16;
typedef __attribute__((ext_vector_type(8)))  float  f32x8;

// ---------------------------------------------------------------------------
// GEMM: C[M_PAD x NCOL] = A[M x 128] * W[128 x NCOL]
// A,W fp32 in memory; bf16 WMMA, fp32 accumulate. 8 waves/block, 16 rows/wave.
// W staged in LDS in *fragment-major* order so each B fragment is one
// 32-byte contiguous vector load per lane (2x ds_load_b128).
// C must have >= gridDim.x*128 rows (padded scratch) -> unconditional stores.
// ---------------------------------------------------------------------------
template<int NCOL>
__global__ __launch_bounds__(256)
void gemm_k128_bf16(const float* __restrict__ A, const float* __restrict__ W,
                    float* __restrict__ C, int M) {
  constexpr int NFRAG = (NCOL / 16) * 4 * 32;   // col-tiles * k-chunks * lanes
  __shared__ bf16x16 sWf[NFRAG];
  __bf16* sflat = (__bf16*)sWf;
  // stage W -> LDS, fragment-major: [((ct*4+kc)*32+lane)*16 + e]
  for (int idx = threadIdx.x; idx < NFRAG * 16; idx += 256) {
    const int e    = idx & 15;
    const int lane = (idx >> 4) & 31;
    const int kc   = (idx >> 9) & 3;
    const int ct   = idx >> 11;
    const int half = lane >> 4;
    const int col  = ct * 16 + (lane & 15);
    const int krow = kc * 32 + half * 8 + (e & 7) + ((e >> 3) << 4);
    sflat[idx] = (__bf16)W[krow * NCOL + col];
  }
  __syncthreads();

  const int wave = threadIdx.x >> 5;
  const int lane = threadIdx.x & 31;
  const int r    = lane & 15;
  const int half = lane >> 4;
  const int row0 = blockIdx.x * 128 + wave * 16;

  int arow = row0 + r;
  if (arow > M - 1) arow = M - 1;      // clamp loads (inputs may be unpadded)
  const float4* Ar4 = (const float4*)(A + (size_t)arow * 128);

  // A fragments, 16-bit A 16x32 layout; explicit float4 loads + pk cvt
  bf16x16 af[4];
#pragma unroll
  for (int kc = 0; kc < 4; ++kc) {
    const int kb4 = kc * 8 + half * 2;            // (kc*32 + half*8)/4
    float4 p0 = Ar4[kb4],     p1 = Ar4[kb4 + 1];  // K = kb .. kb+7
    float4 q0 = Ar4[kb4 + 4], q1 = Ar4[kb4 + 5];  // K = kb+16 .. kb+23
    af[kc][0]  = (__bf16)p0.x; af[kc][1]  = (__bf16)p0.y;
    af[kc][2]  = (__bf16)p0.z; af[kc][3]  = (__bf16)p0.w;
    af[kc][4]  = (__bf16)p1.x; af[kc][5]  = (__bf16)p1.y;
    af[kc][6]  = (__bf16)p1.z; af[kc][7]  = (__bf16)p1.w;
    af[kc][8]  = (__bf16)q0.x; af[kc][9]  = (__bf16)q0.y;
    af[kc][10] = (__bf16)q0.z; af[kc][11] = (__bf16)q0.w;
    af[kc][12] = (__bf16)q1.x; af[kc][13] = (__bf16)q1.y;
    af[kc][14] = (__bf16)q1.z; af[kc][15] = (__bf16)q1.w;
  }

#pragma unroll
  for (int ct = 0; ct < NCOL / 16; ++ct) {
    f32x8 acc = {0.f, 0.f, 0.f, 0.f, 0.f, 0.f, 0.f, 0.f};
#pragma unroll
    for (int kc = 0; kc < 4; ++kc) {
      bf16x16 bf = sWf[(ct * 4 + kc) * 32 + lane];
      acc = __builtin_amdgcn_wmma_f32_16x16x32_bf16(
          false, af[kc], false, bf, (short)0, acc, false, false);
    }
    // C/D layout: VGPR j -> row (8*half + j); lanes 0..15 -> col. Unconditional.
    float* Cp = C + (size_t)(row0 + half * 8) * NCOL + ct * 16 + r;
#pragma unroll
    for (int j = 0; j < 8; ++j) Cp[(size_t)j * NCOL] = acc[j];
  }
}

// wv[i] = sum_h W[i][h] * a[h]   (fold W_dst @ a_dst -> 128-vector)
__global__ void fold_wa(const float* __restrict__ W, const float* __restrict__ a,
                        float* __restrict__ wv, int rows, int cols) {
  int i = blockIdx.x * blockDim.x + threadIdx.x;
  if (i >= rows) return;
  const float* wr = W + (size_t)i * cols;
  float s = 0.f;
  for (int h = 0; h < cols; ++h) s += wr[h] * a[h];
  wv[i] = s;
}

// out[row] = dot(Mtx[row, 0:C], v)  -- one wave per row
__global__ __launch_bounds__(256)
void rowdot(const float* __restrict__ Mtx, const float* __restrict__ v,
            float* __restrict__ out, int n, int C) {
  int row  = blockIdx.x * 8 + (threadIdx.x >> 5);
  int lane = threadIdx.x & 31;
  if (row >= n) return;
  const float* p = Mtx + (size_t)row * C;
  float s = 0.f;
  for (int c = lane; c < C; c += 32) s += p[c] * v[c];
  for (int off = 16; off > 0; off >>= 1) s += __shfl_xor(s, off, 32);
  if (lane == 0) out[row] = s;
}

__global__ void fill_f32(float* __restrict__ p, float v, long long n) {
  long long i = (long long)blockIdx.x * blockDim.x + threadIdx.x;
  if (i < n) p[i] = v;
}

// out[i] = bias[i % C]  (bias broadcast as accumulation base)
__global__ void fill_bias(float* __restrict__ out, const float* __restrict__ bias,
                          long long total, int C) {
  long long i = (long long)blockIdx.x * blockDim.x + threadIdx.x;
  if (i >= total) return;
  out[i] = bias[(int)(i % C)];
}

__global__ void relu_k(float* __restrict__ p, long long n) {
  long long i = (long long)blockIdx.x * blockDim.x + threadIdx.x;
  if (i < n) { float v = p[i]; p[i] = v > 0.f ? v : 0.f; }
}

__device__ inline void atomicMaxFloat(float* addr, float val) {
  // sign-split monotone mapping: works with init value -inf
  if (val >= 0.f) atomicMax((int*)addr, __float_as_int(val));
  else            atomicMin((unsigned int*)addr, __float_as_uint(val));
}

__global__ void edge_logits_max(const int* __restrict__ src, const int* __restrict__ dst,
                                const float* __restrict__ als, const float* __restrict__ ald,
                                float* __restrict__ e, float* __restrict__ m, int E) {
  int i = blockIdx.x * blockDim.x + threadIdx.x;
  if (i >= E) return;
  float x = als[src[i]] + ald[dst[i]];
  x = (x > 0.f) ? x : NEG_SLOPE * x;      // leaky_relu
  e[i] = x;
  atomicMaxFloat(&m[dst[i]], x);
}

__global__ void edge_exp_sum(const int* __restrict__ dst, const float* __restrict__ e,
                             const float* __restrict__ m, float* __restrict__ ex,
                             float* __restrict__ den, int E) {
  int i = blockIdx.x * blockDim.x + threadIdx.x;
  if (i >= E) return;
  int d = dst[i];
  float v = expf(e[i] - m[d]);
  ex[i] = v;
  atomicAdd(&den[d], v);
}

// out[dst] += hs[src] * (ex/den[dst]); one thread per (edge, 4 channels)
__global__ void scatter_msg(const int* __restrict__ src, const int* __restrict__ dst,
                            const float* __restrict__ hs, const float* __restrict__ ex,
                            const float* __restrict__ den, float* __restrict__ out,
                            int E, int C) {
  long long gid = (long long)blockIdx.x * blockDim.x + threadIdx.x;
  int groups = C >> 2;
  long long e = gid / groups;
  int g = (int)(gid - e * groups);
  if (e >= E) return;
  int s = src[e], d = dst[e];
  float alpha = ex[e] / den[d];
  float4 h4 = ((const float4*)(hs + (size_t)s * C))[g];
  float* op = out + (size_t)d * C + (g << 2);
  atomicAdd(op + 0, h4.x * alpha);
  atomicAdd(op + 1, h4.y * alpha);
  atomicAdd(op + 2, h4.z * alpha);
  atomicAdd(op + 3, h4.w * alpha);
}

// pred[e] = sum_c osrc[src[e]][c] * relw[c] * odst[dst[e]][c]   (C = 64)
__global__ __launch_bounds__(256)
void score_k(const int* __restrict__ src, const int* __restrict__ dst,
             const float* __restrict__ osrc, const float* __restrict__ odst,
             const float* __restrict__ relw, float* __restrict__ pred, int E) {
  int e    = blockIdx.x * 8 + (threadIdx.x >> 5);
  int lane = threadIdx.x & 31;
  if (e >= E) return;
  const float* ps = osrc + (size_t)src[e] * OUT_C;
  const float* pd = odst + (size_t)dst[e] * OUT_C;
  float s = ps[lane] * relw[lane] * pd[lane]
          + ps[lane + 32] * relw[lane + 32] * pd[lane + 32];
  for (int off = 16; off > 0; off >>= 1) s += __shfl_xor(s, off, 32);
  if (lane == 0) pred[e] = s;
}

// ---------------------------------------------------------------------------
struct Scratch { float *hs, *als, *ald, *m, *den, *e, *ex, *wv; };

static void run_gat(const float* x_src, const float* x_dst,
                    const int* src, const int* dst,
                    const float* Wsrc, const float* Wdst,
                    const float* asrc, const float* adst,
                    const float* bias, int relu, int Cout,
                    const Scratch& s, float* out, hipStream_t stream) {
  const int N = N_NODES, E = NEDGE;
  fold_wa<<<1, 128, 0, stream>>>(Wdst, adst, s.wv, 128, Cout);
  if (Cout == 128)
    gemm_k128_bf16<128><<<M_PAD / 128, 256, 0, stream>>>(x_src, Wsrc, s.hs, N);
  else
    gemm_k128_bf16<64><<<M_PAD / 128, 256, 0, stream>>>(x_src, Wsrc, s.hs, N);
  rowdot<<<(N + 7) / 8, 256, 0, stream>>>(s.hs, asrc, s.als, N, Cout);   // al_s
  rowdot<<<(N + 7) / 8, 256, 0, stream>>>(x_dst, s.wv, s.ald, N, 128);   // al_d (folded)
  fill_f32<<<(N + 255) / 256, 256, 0, stream>>>(s.m, -INFINITY, N);
  fill_f32<<<(N + 255) / 256, 256, 0, stream>>>(s.den, 0.f, N);
  long long oc = (long long)N * Cout;
  fill_bias<<<(int)((oc + 255) / 256), 256, 0, stream>>>(out, bias, oc, Cout);
  edge_logits_max<<<(E + 255) / 256, 256, 0, stream>>>(src, dst, s.als, s.ald, s.e, s.m, E);
  edge_exp_sum<<<(E + 255) / 256, 256, 0, stream>>>(dst, s.e, s.m, s.ex, s.den, E);
  long long tot = (long long)E * (Cout / 4);
  scatter_msg<<<(int)((tot + 255) / 256), 256, 0, stream>>>(src, dst, s.hs, s.ex, s.den, out, E, Cout);
  if (relu)
    relu_k<<<(int)((oc + 255) / 256), 256, 0, stream>>>(out, oc);
}

extern "C" void kernel_launch(void* const* d_in, const int* in_sizes, int n_in,
                              void* d_out, int out_size, void* d_ws, size_t ws_size,
                              hipStream_t stream) {
  const float* x_a     = (const float*)d_in[0];
  const float* x_b     = (const float*)d_in[1];
  const int*   ei_ab   = (const int*)d_in[2];   // [2, E]
  const int*   ei_ba   = (const int*)d_in[3];
  const float* c1_Wsrc = (const float*)d_in[4]; // [2,128,128]
  const float* c1_Wdst = (const float*)d_in[5];
  const float* c1_asrc = (const float*)d_in[6]; // [2,128]
  const float* c1_adst = (const float*)d_in[7];
  const float* c1_bias = (const float*)d_in[8];
  const float* c2_Wsrc = (const float*)d_in[9];  // [2,128,64]
  const float* c2_Wdst = (const float*)d_in[10];
  const float* c2_asrc = (const float*)d_in[11]; // [2,64]
  const float* c2_adst = (const float*)d_in[12];
  const float* c2_bias = (const float*)d_in[13];
  const float* rel_w   = (const float*)d_in[14]; // [2,64]
  float* out = (float*)d_out;

  const int N = N_NODES;
  float* ws   = (float*)d_ws;
  float* h_a  = ws;                       // N*128
  float* h_b  = h_a  + (size_t)N * 128;   // N*128
  float* o_a  = h_b  + (size_t)N * 128;   // N*64
  float* o_b  = o_a  + (size_t)N * 64;    // N*64
  Scratch s;
  s.hs  = o_b  + (size_t)N * 64;          // M_PAD*128 (padded, reused per GAT)
  s.als = s.hs + (size_t)M_PAD * 128;     // N
  s.ald = s.als + N;                      // N
  s.m   = s.ald + N;                      // N
  s.den = s.m   + N;                      // N
  s.e   = s.den + N;                      // E
  s.ex  = s.e   + NEDGE;                  // E
  s.wv  = s.ex  + NEDGE;                  // 128

  const int* ab_src = ei_ab,  *ab_dst = ei_ab + NEDGE;
  const int* ba_src = ei_ba,  *ba_dst = ei_ba + NEDGE;

  // conv1 (inputs are x only -> relu-in-epilogue is exact)
  run_gat(x_a, x_b, ab_src, ab_dst, c1_Wsrc, c1_Wdst, c1_asrc, c1_adst,
          c1_bias, /*relu=*/1, HID_C, s, h_b, stream);
  run_gat(x_b, x_a, ba_src, ba_dst, c1_Wsrc + D_IN * HID_C, c1_Wdst + D_IN * HID_C,
          c1_asrc + HID_C, c1_adst + HID_C, c1_bias + HID_C, 1, HID_C, s, h_a, stream);
  // conv2 (no activation; bias is pre-filled, so no extra epilogue pass)
  run_gat(h_a, h_b, ab_src, ab_dst, c2_Wsrc, c2_Wdst, c2_asrc, c2_adst,
          c2_bias, 0, OUT_C, s, o_b, stream);
  run_gat(h_b, h_a, ba_src, ba_dst, c2_Wsrc + HID_C * OUT_C, c2_Wdst + HID_C * OUT_C,
          c2_asrc + OUT_C, c2_adst + OUT_C, c2_bias + OUT_C, 0, OUT_C, s, o_a, stream);
  // scoring: out = [pred_ab | pred_ba]
  score_k<<<(NEDGE + 7) / 8, 256, 0, stream>>>(ab_src, ab_dst, o_a, o_b, rel_w, out, NEDGE);
  score_k<<<(NEDGE + 7) / 8, 256, 0, stream>>>(ba_src, ba_dst, o_b, o_a, rel_w + OUT_C, out + NEDGE, NEDGE);
}
